// ScaledDotProductAttention_12051678233304
// MI455X (gfx1250) — matrix operand order repaired and checked by reading, not compiled, not run on hardware
//
#include <hip/hip_runtime.h>

typedef __attribute__((ext_vector_type(16))) __bf16 v16bf;
typedef __attribute__((ext_vector_type(8)))  __bf16 v8bf;
typedef __attribute__((ext_vector_type(8)))  float  v8f;
typedef __attribute__((ext_vector_type(4)))  float  v4f;

#define B_    8
#define TQ    2048
#define TV    2048
#define DD    128
#define BR    128      // query rows per workgroup
#define BC    64       // keys per iteration
#define NWAVE 8

#define NEGBIG (-1.442695e9f)            // -1e9 * log2(e): mask bias in exp2 domain

__global__ __launch_bounds__(256)
void fa_fwd_kernel(const float* __restrict__ Q,
                   const float* __restrict__ V,
                   const float* __restrict__ K,
                   const unsigned char* __restrict__ qmask,
                   const unsigned char* __restrict__ vmask,
                   float* __restrict__ Out)
{
    __shared__ __bf16 ldsK [BC * DD];          // [key][d]  K row-major (bf16)
    __shared__ __bf16 ldsVt[DD * BC];          // [d][key]  V transposed (bf16)
    __shared__ float  ldsBias[BC];             // value-mask bias per key (exp2 domain)
    __shared__ __bf16 ldsP[NWAVE * BC * 16];   // per-wave P^T staging [key][query]

    const int t     = threadIdx.x;
    const int wid   = t >> 5;
    const int lane  = t & 31;
    const int l     = lane & 15;
    const int h     = lane >> 4;
    const int b     = blockIdx.y;
    const int q0    = blockIdx.x * BR;
    const int qbase = q0 + wid * 16;
    const int qr    = qbase + l;               // this lane's query row

    // ---- Q^T B-fragments, pre-scaled by (1/sqrt(D))*log2(e) for exp2 softmax ----
    const float scale2 = 0.08838834764831845f * 1.4426950408889634f;
    v16bf qT[4];
    {
        const float* qb = Q + ((size_t)b * TQ + (size_t)qbase) * DD;
        #pragma unroll
        for (int kk = 0; kk < 4; ++kk)
            #pragma unroll
            for (int j = 0; j < 16; ++j)       // B-layout: col = query l, element j -> d in table order
                qT[kk][j] = (__bf16)(qb[(size_t)l * DD + kk * 32 + 8 * h + (j & 7) + 16 * (j >> 3)] * scale2);
    }

    const v8f vzero = {};
    v8f acc[8];                                // context^T: acc[mm] = d rows 16mm+r+8h, col = query l
    #pragma unroll
    for (int i = 0; i < 8; ++i) acc[i] = vzero;
    float m_run = -__builtin_inff();           // running max (base-2 domain)
    float l_run = 0.f;

    const int kb_max = q0 + BR;                // causal bound for the whole block

    for (int kb = 0; kb < kb_max; kb += BC) {
        __syncthreads();
        // ---- cooperative tile load (fp32 -> bf16): K row-major, V^T into LDS ----
        {
            const int key = t >> 2;            // 0..63
            const int d0  = (t & 3) * 32;      // 0,32,64,96
            const float* kp = K + ((size_t)b * TV + (size_t)(kb + key)) * DD + d0;
            const float* vp = V + ((size_t)b * TV + (size_t)(kb + key)) * DD + d0;
            #pragma unroll
            for (int g = 0; g < 4; ++g) {
                v8bf kc;
                #pragma unroll
                for (int i = 0; i < 8; ++i) kc[i] = (__bf16)kp[g * 8 + i];
                *(v8bf*)&ldsK[key * DD + d0 + g * 8] = kc;
            }
            float vv[32];
            #pragma unroll
            for (int i = 0; i < 32; ++i) vv[i] = vp[i];
            #pragma unroll
            for (int i = 0; i < 32; ++i)
                ldsVt[(d0 + i) * BC + key] = (__bf16)vv[i];
            if (t < BC)
                ldsBias[t] = vmask[(size_t)b * TV + kb + t] ? 0.f : NEGBIG;
            if (kb + BC < kb_max) {            // global_prefetch_b8 next tile
                __builtin_prefetch(kp + (size_t)BC * DD, 0, 0);
                __builtin_prefetch(vp + (size_t)BC * DD, 0, 0);
            }
        }
        __syncthreads();

        // Wave-uniform causal skip: whole 64-key block above this wave's rows.
        if (kb <= qbase + 15) {
            // ---- S^T = K * Q^T : 4 tiles of 16 keys; col = query l ----
            v8f c[4];
            #pragma unroll
            for (int n = 0; n < 4; ++n) c[n] = vzero;
            #pragma unroll
            for (int kk = 0; kk < 4; ++kk) {
                #pragma unroll
                for (int n = 0; n < 4; ++n) {
                    const __bf16* kr = &ldsK[(16 * n + l) * DD + kk * 32];
                    v8bf alo = *(const v8bf*)&kr[8 * h];
                    v8bf ahi = *(const v8bf*)&kr[16 + 8 * h];
                    v16bf a;
                    #pragma unroll
                    for (int i = 0; i < 8; ++i) { a[i] = alo[i]; a[8 + i] = ahi[i]; }
                    c[n] = __builtin_amdgcn_wmma_f32_16x16x32_bf16(false, a, false, qT[kk],
                                                                   (short)0, c[n], false, false);
                }
            }

            // ---- masked online softmax (exp2 domain): keys in-lane ----
            float bias[4][8];
            #pragma unroll
            for (int n = 0; n < 4; ++n) {
                v4f blo = *(const v4f*)&ldsBias[16 * n + 8 * h];
                v4f bhi = *(const v4f*)&ldsBias[16 * n + 8 * h + 4];
                #pragma unroll
                for (int i = 0; i < 4; ++i) { bias[n][i] = blo[i]; bias[n][4 + i] = bhi[i]; }
            }
            const bool diag = (kb + BC > qbase + 1);   // causal test needed?
            float s[4][8];
            float tmax = -__builtin_inff();
            #pragma unroll
            for (int n = 0; n < 4; ++n)
                #pragma unroll
                for (int r = 0; r < 8; ++r) {
                    float mb = bias[n][r];
                    if (diag) {
                        const int kc = kb + 16 * n + r + 8 * h;
                        mb = fminf(mb, (kc <= qr) ? 0.f : NEGBIG);
                    }
                    s[n][r] = c[n][r] + mb;
                    tmax    = fmaxf(tmax, s[n][r]);
                }
            tmax = fmaxf(tmax, __shfl_xor(tmax, 16, 32));   // merge half-wave key groups
            const float mnew  = fmaxf(m_run, tmax);
            const float alpha = __builtin_amdgcn_exp2f(m_run - mnew);
            float psum = 0.f;
            #pragma unroll
            for (int n = 0; n < 4; ++n)
                #pragma unroll
                for (int r = 0; r < 8; ++r) {
                    s[n][r] = __builtin_amdgcn_exp2f(s[n][r] - mnew);  // s becomes p
                    psum   += s[n][r];
                }
            psum  = psum + __shfl_xor(psum, 16, 32);
            l_run = l_run * alpha + psum;
            m_run = mnew;
            #pragma unroll
            for (int mm = 0; mm < 8; ++mm) acc[mm] *= alpha;

            // ---- relayout P^T: C-layout -> B-layout via per-wave LDS staging ----
            {
                __bf16* pp = &ldsP[wid * (BC * 16)];
                asm volatile("" ::: "memory");
                #pragma unroll
                for (int n = 0; n < 4; ++n)
                    #pragma unroll
                    for (int r = 0; r < 8; ++r)
                        pp[(16 * n + r + 8 * h) * 16 + l] = (__bf16)s[n][r];
                // intra-wave cross-lane LDS RAW: DS in-order per wave; fence
                // the compiler and drain DScnt before the B-layout reads.
                asm volatile("s_wait_dscnt 0x0" ::: "memory");
            }
            v16bf pT0, pT1;
            #pragma unroll
            for (int s2 = 0; s2 < 16; ++s2) {
                const int kx = 8 * h + (s2 & 7) + 16 * (s2 >> 3);
                pT0[s2] = ldsP[wid * (BC * 16) + kx * 16 + l];
                pT1[s2] = ldsP[wid * (BC * 16) + (32 + kx) * 16 + l];
            }

            // ---- context^T += V^T * P^T  (8 d-row tiles x 2 key-chunks of 32) ----
            #pragma unroll
            for (int mm = 0; mm < 8; ++mm) {
                const __bf16* vr = &ldsVt[(16 * mm + l) * BC];
                v8bf v0lo = *(const v8bf*)&vr[8 * h];
                v8bf v0hi = *(const v8bf*)&vr[16 + 8 * h];
                v8bf v1lo = *(const v8bf*)&vr[32 + 8 * h];
                v8bf v1hi = *(const v8bf*)&vr[48 + 8 * h];
                v16bf va0, va1;
                #pragma unroll
                for (int i = 0; i < 8; ++i) {
                    va0[i] = v0lo[i]; va0[8 + i] = v0hi[i];
                    va1[i] = v1lo[i]; va1[8 + i] = v1hi[i];
                }
                acc[mm] = __builtin_amdgcn_wmma_f32_16x16x32_bf16(false, va0, false, pT0,
                                                                  (short)0, acc[mm], false, false);
                acc[mm] = __builtin_amdgcn_wmma_f32_16x16x32_bf16(false, va1, false, pT1,
                                                                  (short)0, acc[mm], false, false);
            }
        }
    }

    // ---- epilogue: normalize, apply query mask, vectorized row store ----
    const float qm  = qmask[(size_t)b * TQ + qr] ? 1.f : 0.f;
    const float inv = (l_run > 0.f) ? (qm / l_run) : 0.f;
    float* op = Out + ((size_t)b * TQ + (size_t)qr) * DD;
    #pragma unroll
    for (int mm = 0; mm < 8; ++mm) {
        v4f o0, o1;
        #pragma unroll
        for (int i = 0; i < 4; ++i) { o0[i] = acc[mm][i] * inv; o1[i] = acc[mm][4 + i] * inv; }
        *(v4f*)&op[16 * mm + 8 * h]     = o0;
        *(v4f*)&op[16 * mm + 8 * h + 4] = o1;
    }
}

extern "C" void kernel_launch(void* const* d_in, const int* in_sizes, int n_in,
                              void* d_out, int out_size, void* d_ws, size_t ws_size,
                              hipStream_t stream) {
    (void)in_sizes; (void)n_in; (void)out_size; (void)d_ws; (void)ws_size;
    const float*         Q  = (const float*)d_in[0];
    const float*         V  = (const float*)d_in[1];   // setup_inputs order: query, value, key
    const float*         K  = (const float*)d_in[2];
    const unsigned char* qm = (const unsigned char*)d_in[3];
    const unsigned char* vm = (const unsigned char*)d_in[4];
    float* Out = (float*)d_out;

    dim3 grid(TQ / BR, B_);
    dim3 block(256);
    fa_fwd_kernel<<<grid, block, 0, stream>>>(Q, V, K, qm, vm, Out);
}
